// LearnedGraphModule_7456063226580
// MI455X (gfx1250) — matrix operand order repaired and checked by reading, not compiled, not run on hardware
//
#include <hip/hip_runtime.h>
#include <math.h>

// ---------------------------------------------------------------------------
// LearnedGraphModule on MI455X (gfx1250, wave32, WMMA 16x16x32 f16)
//
//   k1: ZZ = nodes @ [W1a; W1b; Wm]^T  (B,N,192) via WMMA -> ZS | ZT | MMR
//       (MMR gets +bm and relu folded in; relu commutes with the gather).
//   k2: per 8-node block (192 edges):
//       H1(192x64 f16, LDS) = relu(ZS[n] + gather(ZT) + zpos[k] + b1)
//       H2(192x32 f16)      = relu(H1 @ W2^T + b2)       <- WMMA (24 jobs/8 waves)
//       scores              = sigmoid(H2 @ w3 + b3)
//       mask (thr / top8 / top3, jax tie order), weights = s * keep * mask
//       agg                 = sum_k w * gather(MMR) / (sum w + eps)
//       out                 = x + agg @ Wo^T + bo         <- WMMA, NCHW store
// ---------------------------------------------------------------------------

typedef __attribute__((ext_vector_type(16))) _Float16 v16h;
typedef __attribute__((ext_vector_type(8)))  float    v8f;

#define BATCH 2
#define CCH   64
#define HH    128
#define WW    128
#define NN    (HH*WW)
#define EIN   130
#define KNB   24
#define MAXE  8
#define MINE  3
#define TEMP_ 0.1f
#define EPS_  1e-6f

__device__ __forceinline__ float sigmoidf_(float x){ return 1.0f/(1.0f+expf(-x)); }

// A fragment (16x32 f16) from row-major LDS [rows][stride] at (row0, kbase).
// ISA 7.12.2: lanes 0-15: row=lane, K in {0..7,16..23};
//             lanes 16-31: row=lane-16, K in {8..15,24..31}.
__device__ __forceinline__ v16h load_a16x32(const _Float16* base, int row0, int stride,
                                            int kbase, int lane){
  int r    = lane & 15;
  int koff = (lane < 16) ? 0 : 8;
  const _Float16* p = base + (row0 + r)*stride + kbase + koff;
  v16h a;
  #pragma unroll
  for (int h = 0; h < 8; ++h)  a[h] = p[h];        // K = koff + h
  #pragma unroll
  for (int h = 8; h < 16; ++h) a[h] = p[h + 8];    // K = koff + h + 8
  return a;
}

// B fragment (32x16 f16); source is a row-major weight matrix W[n][k]
// (stride = K length), i.e. B(k,n) = W[ncol0+n][kbase+k].
// Lane = N column (lane&15); lanes 0-15 hold K={0..7,16..23}, lanes 16-31 K={8..15,24..31}.
__device__ __forceinline__ v16h load_bT(const _Float16* base, int ncol0, int stride,
                                        int kbase, int lane){
  int n  = lane & 15;
  int kk = kbase + ((lane < 16) ? 0 : 8);
  const _Float16* p = base + (ncol0 + n)*stride + kk;
  v16h b;
  #pragma unroll
  for (int h = 0; h < 8; ++h)  b[h] = p[h];
  #pragma unroll
  for (int h = 8; h < 16; ++h) b[h] = p[h + 8];
  return b;
}

// ---------------------------------------------------------------------------
// Kernel 1: fused node GEMM. 64 nodes/block, 8 waves, 48 output tiles.
// ---------------------------------------------------------------------------
__global__ __launch_bounds__(256) void k1_precompute(
    const float* __restrict__ x,  const float* __restrict__ W1,
    const float* __restrict__ Wm, const float* __restrict__ bm,
    float* __restrict__ ZS, float* __restrict__ ZT, float* __restrict__ MMR)
{
  __shared__ _Float16 Ax[64*64];    // node tile, [node][ch]
  __shared__ _Float16 Wc[192*64];   // [W1a; W1b; Wm], [outch][inch]
  __shared__ float    bms[64];

  const int tid = threadIdx.x;
  const int b   = blockIdx.x >> 8;
  const int n0  = (blockIdx.x & 255) * 64;

  // x is (B,C,N): consecutive tid -> consecutive node -> coalesced.
  for (int idx = tid; idx < 64*64; idx += 256){
    int node = idx & 63, c = idx >> 6;
    Ax[node*64 + c] = (_Float16)x[((size_t)b*CCH + c)*NN + n0 + node];
  }
  for (int idx = tid; idx < 192*64; idx += 256){
    int r = idx >> 6, c = idx & 63;
    float v;
    if      (r < 64)  v = W1[r*EIN + c];            // z_src weights
    else if (r < 128) v = W1[(r-64)*EIN + 64 + c];  // z_tgt weights
    else              v = Wm[(r-128)*64 + c];       // message weights
    Wc[idx] = (_Float16)v;
  }
  if (tid < 64) bms[tid] = bm[tid];
  __syncthreads();

  // wave is wave-uniform: pin it to an SGPR so region branches go scalar.
  const int wave = __builtin_amdgcn_readfirstlane(tid >> 5);
  const int lane = tid & 31;
  const int chl  = lane & 15;
  const int rbase = (lane < 16) ? 0 : 8;

  #pragma unroll
  for (int t = 0; t < 6; ++t){
    int j  = wave*6 + t;
    int mi = j & 3;          // node tile (0..3)
    int oi = j >> 2;         // out-channel tile (0..11), wave-uniform
    v8f acc = {};
    #pragma unroll
    for (int ks = 0; ks < 2; ++ks){
      v16h a  = load_a16x32(Ax, mi*16, 64, ks*32, lane);
      v16h bb = load_bT   (Wc, oi*16, 64, ks*32, lane);
      acc = __builtin_amdgcn_wmma_f32_16x16x32_f16(false, a, false, bb,
                                                   (short)0, acc, false, false);
    }
    size_t rowbase = ((size_t)b*NN + n0 + mi*16 + rbase)*64;
    if (oi < 4){                       // scalar branch: ZS region
      int ch = oi*16 + chl;
      #pragma unroll
      for (int r = 0; r < 8; ++r) ZS[rowbase + (size_t)r*64 + ch] = acc[r];
    } else if (oi < 8){                // ZT region
      int ch = (oi - 4)*16 + chl;
      #pragma unroll
      for (int r = 0; r < 8; ++r) ZT[rowbase + (size_t)r*64 + ch] = acc[r];
    } else {                           // MMR region: +bm, relu
      int ch = (oi - 8)*16 + chl;
      float bv = bms[ch];
      #pragma unroll
      for (int r = 0; r < 8; ++r)
        MMR[rowbase + (size_t)r*64 + ch] = fmaxf(acc[r] + bv, 0.0f);
    }
  }
}

// ---------------------------------------------------------------------------
// Kernel 2: fused edge scorer + mask + aggregation + output GEMM.
// 8 nodes (192 edges) per block, 8 waves, 24 balanced WMMA jobs.
// ---------------------------------------------------------------------------
__global__ __launch_bounds__(256) void k2_edges(
    const float* __restrict__ x,  const float* __restrict__ W1,
    const float* __restrict__ b1, const float* __restrict__ W2,
    const float* __restrict__ b2, const float* __restrict__ W3,
    const float* __restrict__ b3, const float* __restrict__ thr,
    const float* __restrict__ Wo, const float* __restrict__ bo,
    const float* __restrict__ ZS, const float* __restrict__ ZT,
    const float* __restrict__ MMR, float* __restrict__ out)
{
  __shared__ float    zpb[KNB*64];   // z_pos[k] + b1 folded      (6 KB)
  __shared__ _Float16 W2h[32*64];    // W2 row-major (B source)   (4 KB)
  __shared__ _Float16 Woh[64*64];    // Wo row-major              (8 KB)
  __shared__ _Float16 H1[192*64];    // edges x hid, A operand    (24 KB)
  __shared__ _Float16 H2[192*32];    // relu(h1 W2^T + b2), f16   (12 KB)
  __shared__ float    sc[192], wgt[192], wsum[8];
  __shared__ float    aggf[8*64];
  __shared__ _Float16 aggh[16*64];   // padded A operand for output GEMM
  __shared__ float    b2s[32], w3s[32], bos[64];
  __shared__ float    b3s, thrv;

  const int tid = threadIdx.x;
  const int b   = blockIdx.x >> 11;            // 2048 groups per batch
  const int n0  = (blockIdx.x & 2047) * 8;

  for (int idx = tid; idx < KNB*64; idx += 256){
    int k = idx >> 6, j = idx & 63;
    int kk = (k < 12) ? k : k + 1;             // skip (0,0) in 5x5
    float dy = (float)(kk/5 - 2), dx = (float)(kk%5 - 2);
    zpb[idx] = 0.5f*dx*W1[j*EIN + 128] + 0.5f*dy*W1[j*EIN + 129] + b1[j];
  }
  for (int idx = tid; idx < 32*64; idx += 256) W2h[idx] = (_Float16)W2[idx];
  for (int idx = tid; idx < 64*64; idx += 256) Woh[idx] = (_Float16)Wo[idx];
  if (tid < 32){ b2s[tid] = b2[tid]; w3s[tid] = W3[tid]; }
  if (tid < 64) bos[tid] = bo[tid];
  if (tid == 0){ b3s = b3[0]; thrv = sigmoidf_(thr[0]); }
  __syncthreads();

  // Build H1: gather ZT rows (L2-resident), add z_src + zpos + b1, relu -> f16.
  for (int idx = tid; idx < 192*64; idx += 256){
    int e = idx >> 6, j = idx & 63;
    int nl = e / KNB, k = e % KNB;
    int n = n0 + nl;
    int y = n >> 7, xc = n & 127;
    int kk = (k < 12) ? k : k + 1;
    int dy = kk/5 - 2, dx = kk%5 - 2;
    int ny = min(max(y + dy, 0), HH-1), nx = min(max(xc + dx, 0), WW-1);
    int nb = ny*WW + nx;
    float v = ZS[((size_t)b*NN + n )*64 + j]
            + ZT[((size_t)b*NN + nb)*64 + j] + zpb[k*64 + j];
    H1[idx] = (_Float16)fmaxf(v, 0.0f);
  }
  __syncthreads();

  // H2 = relu(H1 @ W2^T + b2): (192x64)@(64x32) = 12 Mtiles x 2 Ntiles.
  const int wave  = __builtin_amdgcn_readfirstlane(tid >> 5);
  const int lane  = tid & 31;
  const int chl   = lane & 15;
  const int rbase = (lane < 16) ? 0 : 8;
  #pragma unroll
  for (int t = 0; t < 3; ++t){                 // 24 jobs = 3 per wave, balanced
    int j  = wave*3 + t;
    int mt = j >> 1, nt = j & 1;
    v8f acc = {};
    #pragma unroll
    for (int ks = 0; ks < 2; ++ks){
      v16h a  = load_a16x32(H1,  mt*16, 64, ks*32, lane);
      v16h bb = load_bT   (W2h, nt*16, 64, ks*32, lane);
      acc = __builtin_amdgcn_wmma_f32_16x16x32_f16(false, a, false, bb,
                                                   (short)0, acc, false, false);
    }
    int ch = nt*16 + chl;
    float bv = b2s[ch];
    #pragma unroll
    for (int r = 0; r < 8; ++r){
      int e = mt*16 + rbase + r;
      H2[e*32 + ch] = (_Float16)fmaxf(acc[r] + bv, 0.0f);
    }
  }
  __syncthreads();

  // Edge scores.
  if (tid < 192){
    float a = b3s;
    #pragma unroll
    for (int c = 0; c < 32; ++c) a += (float)H2[tid*32 + c] * w3s[c];
    sc[tid] = sigmoidf_(a);
  }
  __syncthreads();

  // Per-node mask (threshold / top8 / top3, jax tie order) + edge weights.
  if (tid < 8){
    const float* s = sc + tid*KNB;
    float tv = thrv;
    int cnt = 0;
    unsigned mthr = 0;
    #pragma unroll
    for (int k = 0; k < KNB; ++k){
      if (s[k] >= tv){ ++cnt; mthr |= (1u << k); }
    }
    unsigned taken = 0, m3 = 0, m8 = 0;
    for (int t = 0; t < MAXE; ++t){            // ordered top-8 selection
      int best = 0; float bv = -1.0f;
      for (int k = 0; k < KNB; ++k){
        if (taken & (1u << k)) continue;
        if (s[k] > bv){ bv = s[k]; best = k; } // strict > keeps lowest tie index
      }
      taken |= (1u << best);
      m8    |= (1u << best);
      if (t < MINE) m3 |= (1u << best);
    }
    unsigned fm = (cnt > MAXE) ? m8 : ((cnt < MINE) ? m3 : mthr);
    float wsacc = 0.0f;
    for (int k = 0; k < KNB; ++k){
      float keep = sigmoidf_((s[k] - tv) / TEMP_);
      float w = ((fm >> k) & 1u) ? s[k]*keep : 0.0f;
      wgt[tid*KNB + k] = w;
      wsacc += w;
    }
    wsum[tid] = wsacc;
  }
  __syncthreads();

  // Aggregation: (node, channel) pairs; MMR gathers hit L2.
  for (int idx = tid; idx < 8*64; idx += 256){
    int nl = idx >> 6, c = idx & 63;
    int n = n0 + nl;
    int y = n >> 7, xc = n & 127;
    float acc = 0.0f;
    #pragma unroll
    for (int k = 0; k < KNB; ++k){
      float w = wgt[nl*KNB + k];
      int kk = (k < 12) ? k : k + 1;
      int dy = kk/5 - 2, dx = kk%5 - 2;
      int ny = min(max(y + dy, 0), HH-1), nx = min(max(xc + dx, 0), WW-1);
      acc += w * MMR[((size_t)b*NN + ny*WW + nx)*64 + c];
    }
    aggf[nl*64 + c] = acc / (wsum[nl] + EPS_);
  }
  __syncthreads();
  for (int idx = tid; idx < 16*64; idx += 256){
    int r = idx >> 6;
    aggh[idx] = (r < 8) ? (_Float16)aggf[idx] : (_Float16)0.0f;
  }
  __syncthreads();

  // out = x + agg @ Wo^T + bo : waves 0..3 each own one 16-wide channel tile.
  if (wave < 4){                               // scalar branch
    int nt = wave;
    v8f acc = {};
    #pragma unroll
    for (int ks = 0; ks < 2; ++ks){
      v16h a  = load_a16x32(aggh, 0,     64, ks*32, lane);
      v16h bb = load_bT   (Woh,  nt*16, 64, ks*32, lane);
      acc = __builtin_amdgcn_wmma_f32_16x16x32_f16(false, a, false, bb,
                                                   (short)0, acc, false, false);
    }
    // Rows 0..7 of the low half-wave are the 8 real nodes; high half is pad.
    if (lane < 16){
      int ch = nt*16 + chl;
      float bv = bos[ch];
      size_t obase = ((size_t)b*CCH + ch)*NN + n0;   // NCHW layout
      #pragma unroll
      for (int r = 0; r < 8; ++r){
        out[obase + r] = x[obase + r] + acc[r] + bv;
      }
    }
  }
}

// ---------------------------------------------------------------------------
extern "C" void kernel_launch(void* const* d_in, const int* in_sizes, int n_in,
                              void* d_out, int out_size, void* d_ws, size_t ws_size,
                              hipStream_t stream)
{
  (void)in_sizes; (void)n_in; (void)out_size; (void)ws_size;
  const float* x   = (const float*)d_in[0];
  const float* W1  = (const float*)d_in[1];
  const float* b1  = (const float*)d_in[2];
  const float* W2  = (const float*)d_in[3];
  const float* b2  = (const float*)d_in[4];
  const float* W3  = (const float*)d_in[5];
  const float* b3  = (const float*)d_in[6];
  const float* thr = (const float*)d_in[7];
  const float* Wm  = (const float*)d_in[8];
  const float* bm  = (const float*)d_in[9];
  const float* Wo  = (const float*)d_in[10];
  const float* bo  = (const float*)d_in[11];
  float* out = (float*)d_out;

  float* ZS  = (float*)d_ws;                         // (B,N,64)
  float* ZT  = ZS  + (size_t)BATCH*NN*64;            // (B,N,64)
  float* MMR = ZT  + (size_t)BATCH*NN*64;            // (B,N,64), relu(Wm x + bm)

  k1_precompute<<<dim3(BATCH*(NN/64)), dim3(256), 0, stream>>>(
      x, W1, Wm, bm, ZS, ZT, MMR);
  k2_edges<<<dim3(BATCH*(NN/8)), dim3(256), 0, stream>>>(
      x, W1, b1, W2, b2, W3, b3, thr, Wo, bo, ZS, ZT, MMR, out);
}
